// Block_88244398064140
// MI455X (gfx1250) — matrix operand order repaired and checked by reading, not compiled, not run on hardware
//
#include <hip/hip_runtime.h>
#include <hip/hip_bf16.h>
#include <math.h>

// ---------- problem constants ----------
#define BATCH   32
#define SEQ     577
#define CH      1024
#define NHEAD   16
#define HD      64
#define HIDD    4096
#define MTOK    (BATCH * SEQ)     // 18464 token rows
#define NPAD    608               // 577 keys padded to 19*32
#define BH      (BATCH * NHEAD)   // 512

typedef __attribute__((ext_vector_type(16))) __bf16 bf16x16;
typedef __attribute__((ext_vector_type(8)))  float  f32x8;

union FragAB {
  bf16x16 v;
  uint4   q[2];
};

__device__ __forceinline__ __bf16 f2bf(float f) {
  unsigned u = __builtin_bit_cast(unsigned, f);
  u = (u + 0x7FFFu + ((u >> 16) & 1u)) >> 16;   // round-to-nearest-even
  unsigned short s = (unsigned short)u;
  return __builtin_bit_cast(__bf16, s);
}

// ---- gfx1250 async global->LDS path (ASYNCcnt-tracked), with safe fallback ----
// The clang builtin exists on this toolchain but demands LangAS-qualified
// pointer types we cannot spell portably from HIP; use inline asm instead.
// Capability probe: if the builtin is declared, the assembler knows the op.
#if defined(__has_builtin)
#if __has_builtin(__builtin_amdgcn_global_load_async_to_lds_b128)
#define HAVE_ASYNC_LDS 1
#endif
#endif
#ifndef HAVE_ASYNC_LDS
#define HAVE_ASYNC_LDS 0
#endif

#if HAVE_ASYNC_LDS
// vdst = 32-bit LDS byte address (low 32 bits of flat shared-aperture addr),
// vaddr = 64-bit global address in a VGPR pair, saddr = off (NULL).
#define ASYNC_LD_B128(gptr, lptr)                                      \
  asm volatile("global_load_async_to_lds_b128 %0, %1, off"             \
               :: "v"((unsigned)(uintptr_t)(lptr)),                    \
                  "v"((unsigned long long)(uintptr_t)(gptr))           \
               : "memory")
#if __has_builtin(__builtin_amdgcn_s_wait_asynccnt)
#define WAIT_ASYNC(n) __builtin_amdgcn_s_wait_asynccnt(n)
#else
#define WAIT_ASYNC(n) asm volatile("s_wait_asynccnt %0" ::"n"(n) : "memory")
#endif
#endif

// ---------- fp32 -> bf16 cast (weights) ----------
__global__ void cast_bf16_kernel(const float* __restrict__ in,
                                 __bf16* __restrict__ out, long n) {
  long i = (long)blockIdx.x * blockDim.x + threadIdx.x;
  long stride = (long)gridDim.x * blockDim.x;
  for (; i < n; i += stride) out[i] = f2bf(in[i]);
}

// ---------- LayerNorm (+ bf16 cast), one 256-thread block per token ----------
__global__ __launch_bounds__(256)
void ln_bf16_kernel(const float* __restrict__ x, const float* __restrict__ g,
                    const float* __restrict__ b, __bf16* __restrict__ out) {
  __shared__ float red[256];
  const int row = blockIdx.x;
  const int t = threadIdx.x;
  const float* xr = x + (long)row * CH;
  float v[4];
  float s = 0.f;
#pragma unroll
  for (int i = 0; i < 4; ++i) { v[i] = xr[t + 256 * i]; s += v[i]; }
  red[t] = s; __syncthreads();
  for (int off = 128; off > 0; off >>= 1) {
    if (t < off) red[t] += red[t + off];
    __syncthreads();
  }
  const float mu = red[0] * (1.0f / CH);
  __syncthreads();
  s = 0.f;
#pragma unroll
  for (int i = 0; i < 4; ++i) { float d = v[i] - mu; s += d * d; }
  red[t] = s; __syncthreads();
  for (int off = 128; off > 0; off >>= 1) {
    if (t < off) red[t] += red[t + off];
    __syncthreads();
  }
  const float rstd = rsqrtf(red[0] * (1.0f / CH) + 1e-5f);
#pragma unroll
  for (int i = 0; i < 4; ++i) {
    int c = t + 256 * i;
    out[(long)row * CH + c] = f2bf((v[i] - mu) * rstd * g[c] + b[c]);
  }
}

// ---------- generic bf16 GEMM: out[M,Nout] = A[M,K] x W[Nout,K]^T ----------
// mode 0: f32 out (+bias, +residual)      (proj, fc2)
// mode 1: bf16 out, +bias, exact GELU     (fc1)
// mode 2: qkv scatter into q / k / v^T    (qkv)
#define LDT 40  // LDS row stride in bf16 elements (32 + 8 pad)

__global__ __launch_bounds__(256)
void gemm_bf16_kernel(const __bf16* __restrict__ A, const __bf16* __restrict__ W,
                      const float* __restrict__ bias, const float* __restrict__ resid,
                      float* __restrict__ outF, __bf16* __restrict__ outB,
                      __bf16* __restrict__ qOut, __bf16* __restrict__ kOut,
                      __bf16* __restrict__ vtOut,
                      int M, int Nout, int K, int mode) {
#if HAVE_ASYNC_LDS
  __shared__ __align__(16) __bf16 As[2][128 * LDT];
  __shared__ __align__(16) __bf16 Ws[2][128 * LDT];
#else
  __shared__ __align__(16) __bf16 As[1][128 * LDT];
  __shared__ __align__(16) __bf16 Ws[1][128 * LDT];
#endif

  const int tid  = threadIdx.x;
  const int lane = tid & 31;
  const int w    = tid >> 5;          // 8 waves
  const int wr   = w >> 1;            // 0..3 -> 32-row slab
  const int wc   = w & 1;             // 0..1 -> 64-col slab
  const int h    = lane >> 4;         // lane half
  const int ml   = lane & 15;

  const int mBase = blockIdx.y * 128;
  const int nBase = blockIdx.x * 128;

  const int lrow = tid >> 1;          // 0..127
  const int lcol = (tid & 1) * 16;    // 0 or 16 bf16 elements (32B chunks)

  int grA = mBase + lrow; if (grA >= M) grA = M - 1;
  const __bf16* gA = A + (long)grA * K + lcol;
  const __bf16* gW = W + (long)(nBase + lrow) * K + lcol;  // Nout % 128 == 0

  f32x8 acc[2][4];
#pragma unroll
  for (int i = 0; i < 2; ++i)
#pragma unroll
    for (int j = 0; j < 4; ++j)
#pragma unroll
      for (int e = 0; e < 8; ++e) acc[i][j][e] = 0.f;

  const int nSteps = K >> 5;  // K / 32

#if HAVE_ASYNC_LDS
  // -------- double-buffered async pipeline (global -> LDS, no VGPR bounce) ----
  {
    // stage 0
    ASYNC_LD_B128(gA, &As[0][lrow * LDT + lcol]);
    ASYNC_LD_B128(gW, &Ws[0][lrow * LDT + lcol]);
  }
  for (int step = 0; step < nSteps; ++step) {
    const int buf = step & 1;
    if (step + 1 < nSteps) {
      const int k0 = (step + 1) << 5;
      const int nbuf = (step + 1) & 1;
      ASYNC_LD_B128(gA + k0, &As[nbuf][lrow * LDT + lcol]);
      ASYNC_LD_B128(gW + k0, &Ws[nbuf][lrow * LDT + lcol]);
      WAIT_ASYNC(2);   // stage `step`'s 2 copies retired (in-order per wave)
    } else {
      WAIT_ASYNC(0);
    }
    __syncthreads();   // all waves' copies for this stage visible in LDS

    FragAB a[2];
#pragma unroll
    for (int tm = 0; tm < 2; ++tm) {
      const __bf16* ar = &As[buf][(wr * 32 + tm * 16 + ml) * LDT];
      a[tm].q[0] = *(const uint4*)(ar + 8 * h);        // K = 8h .. 8h+7
      a[tm].q[1] = *(const uint4*)(ar + 16 + 8 * h);   // K = 16+8h ..
    }
#pragma unroll
    for (int tn = 0; tn < 4; ++tn) {
      FragAB bfr;
      const __bf16* br = &Ws[buf][(wc * 64 + tn * 16 + ml) * LDT];
      bfr.q[0] = *(const uint4*)(br + 8 * h);          // K = 8h .. 8h+7
      bfr.q[1] = *(const uint4*)(br + 16 + 8 * h);     // K = 16+8h ..
#pragma unroll
      for (int tm = 0; tm < 2; ++tm) {
        acc[tm][tn] = __builtin_amdgcn_wmma_f32_16x16x32_bf16(
            false, a[tm].v, false, bfr.v, (short)0, acc[tm][tn], false, false);
      }
    }
    __syncthreads();   // done reading buf before stage step+2 overwrites it
  }
#else
  // -------- fallback: staged through VGPRs, single buffer -------------------
  for (int step = 0; step < nSteps; ++step) {
    const int k0 = step << 5;
    {
      const uint4* srcA = (const uint4*)(gA + k0);
      uint4* dstA = (uint4*)(&As[0][lrow * LDT + lcol]);
      dstA[0] = srcA[0];
      const uint4* srcW = (const uint4*)(gW + k0);
      uint4* dstW = (uint4*)(&Ws[0][lrow * LDT + lcol]);
      dstW[0] = srcW[0];
    }
    __syncthreads();
    FragAB a[2];
#pragma unroll
    for (int tm = 0; tm < 2; ++tm) {
      const __bf16* ar = &As[0][(wr * 32 + tm * 16 + ml) * LDT];
      a[tm].q[0] = *(const uint4*)(ar + 8 * h);
      a[tm].q[1] = *(const uint4*)(ar + 16 + 8 * h);
    }
#pragma unroll
    for (int tn = 0; tn < 4; ++tn) {
      FragAB bfr;
      const __bf16* br = &Ws[0][(wc * 64 + tn * 16 + ml) * LDT];
      bfr.q[0] = *(const uint4*)(br + 8 * h);
      bfr.q[1] = *(const uint4*)(br + 16 + 8 * h);
#pragma unroll
      for (int tm = 0; tm < 2; ++tm) {
        acc[tm][tn] = __builtin_amdgcn_wmma_f32_16x16x32_bf16(
            false, a[tm].v, false, bfr.v, (short)0, acc[tm][tn], false, false);
      }
    }
    __syncthreads();
  }
#endif

  // epilogue: C/D layout -> element (r + 8h, ml) of each 16x16 tile
#pragma unroll
  for (int tm = 0; tm < 2; ++tm)
#pragma unroll
    for (int tn = 0; tn < 4; ++tn)
#pragma unroll
      for (int r = 0; r < 8; ++r) {
        const int grow = mBase + wr * 32 + tm * 16 + r + 8 * h;
        const int gcol = nBase + wc * 64 + tn * 16 + ml;
        if (grow >= M) continue;
        float val = acc[tm][tn][r];
        if (mode == 0) {
          if (bias)  val += bias[gcol];
          if (resid) val += resid[(long)grow * Nout + gcol];
          outF[(long)grow * Nout + gcol] = val;
        } else if (mode == 1) {
          val += bias[gcol];
          val = 0.5f * val * (1.0f + erff(val * 0.70710678118f));  // exact GELU
          outB[(long)grow * Nout + gcol] = f2bf(val);
        } else {
          // qkv scatter: col = t*1024 + head*64 + d
          const int brow = grow / SEQ, n = grow - brow * SEQ;
          const int tsel = gcol >> 10, hh = (gcol >> 6) & 15, d = gcol & 63;
          const int bh = brow * NHEAD + hh;
          if (tsel == 0)       qOut[((long)bh * SEQ + n) * HD + d] = f2bf(val * 0.125f);
          else if (tsel == 1)  kOut[((long)bh * SEQ + n) * HD + d] = f2bf(val);
          else                 vtOut[((long)bh * HD + d) * NPAD + n] = f2bf(val);
        }
      }
}

// ---------- fused attention: scores -> softmax -> AV, per (b,h,qtile16) ----------
__global__ __launch_bounds__(128)
void attn_kernel(const __bf16* __restrict__ q, const __bf16* __restrict__ k,
                 const __bf16* __restrict__ vt, __bf16* __restrict__ out) {
  __shared__ __align__(16) float  Ssc[16][616];
  __shared__ __align__(16) __bf16 Pb[16][616];

  const int lane = threadIdx.x & 31;
  const int w    = threadIdx.x >> 5;   // 4 waves
  const int h    = lane >> 4;
  const int ml   = lane & 15;
  const int qt   = blockIdx.x;         // 0..36
  const int bh   = blockIdx.y;         // 0..511

  const __bf16* qbase = q  + (long)bh * SEQ * HD;
  const __bf16* kbase = k  + (long)bh * SEQ * HD;
  const __bf16* vbase = vt + (long)bh * HD * NPAD;

  // q A-fragments for the two K=32 steps over D=64 (q already scaled by D^-1/2)
  FragAB aq[2];
  {
    int row = qt * 16 + ml; if (row >= SEQ) row = SEQ - 1;
    const __bf16* qr = qbase + (long)row * HD;
#pragma unroll
    for (int s = 0; s < 2; ++s) {
      aq[s].q[0] = *(const uint4*)(qr + 32 * s + 8 * h);
      aq[s].q[1] = *(const uint4*)(qr + 32 * s + 16 + 8 * h);
    }
  }

  // scores: each wave strides over 16-key tiles
  for (int kt = w; kt < 37; kt += 4) {
    f32x8 c;
#pragma unroll
    for (int e = 0; e < 8; ++e) c[e] = 0.f;
    int key = kt * 16 + ml; if (key >= SEQ) key = SEQ - 1;
    const __bf16* kr = kbase + (long)key * HD;
#pragma unroll
    for (int s = 0; s < 2; ++s) {
      FragAB bfr;
      bfr.q[0] = *(const uint4*)(kr + 32 * s + 8 * h);
      bfr.q[1] = *(const uint4*)(kr + 32 * s + 16 + 8 * h);
      c = __builtin_amdgcn_wmma_f32_16x16x32_bf16(
          false, aq[s].v, false, bfr.v, (short)0, c, false, false);
    }
#pragma unroll
    for (int r = 0; r < 8; ++r)
      Ssc[r + 8 * h][kt * 16 + ml] = c[r];
  }
  __syncthreads();

  // mask padded keys
  for (int col = SEQ + threadIdx.x; col < 616; col += 128)
#pragma unroll
    for (int r = 0; r < 16; ++r) Ssc[r][col] = -1e30f;
  __syncthreads();

  // softmax per row (4 rows per wave), probs -> bf16 LDS
  for (int r = w; r < 16; r += 4) {
    float m = -1e30f;
    for (int col = lane; col < NPAD; col += 32) m = fmaxf(m, Ssc[r][col]);
#pragma unroll
    for (int off = 16; off > 0; off >>= 1) m = fmaxf(m, __shfl_xor(m, off, 32));
    float s = 0.f;
    for (int col = lane; col < NPAD; col += 32) s += __expf(Ssc[r][col] - m);
#pragma unroll
    for (int off = 16; off > 0; off >>= 1) s += __shfl_xor(s, off, 32);
    const float inv = 1.0f / s;
    for (int col = lane; col < NPAD; col += 32)
      Pb[r][col] = f2bf(__expf(Ssc[r][col] - m) * inv);
  }
  __syncthreads();

  // AV: wave w owns d-columns [16w, 16w+16); contract over 608 padded keys
  f32x8 o;
#pragma unroll
  for (int e = 0; e < 8; ++e) o[e] = 0.f;
  const int dt = w;
  for (int s = 0; s < 19; ++s) {
    FragAB a, bfr;
    const __bf16* pr = &Pb[ml][0];
    a.q[0] = *(const uint4*)(pr + 32 * s + 8 * h);
    a.q[1] = *(const uint4*)(pr + 32 * s + 16 + 8 * h);
    const __bf16* vr = vbase + (long)(dt * 16 + ml) * NPAD;
    bfr.q[0] = *(const uint4*)(vr + 32 * s + 8 * h);
    bfr.q[1] = *(const uint4*)(vr + 32 * s + 16 + 8 * h);
    o = __builtin_amdgcn_wmma_f32_16x16x32_bf16(
        false, a.v, false, bfr.v, (short)0, o, false, false);
  }
  const int b_ = bh >> 4, hh = bh & 15;
#pragma unroll
  for (int r = 0; r < 8; ++r) {
    const int row = qt * 16 + r + 8 * h;
    if (row < SEQ)
      out[((long)(b_ * SEQ + row)) * CH + hh * 64 + dt * 16 + ml] = f2bf(o[r]);
  }
}

// ---------- host orchestration ----------
extern "C" void kernel_launch(void* const* d_in, const int* in_sizes, int n_in,
                              void* d_out, int out_size, void* d_ws, size_t ws_size,
                              hipStream_t stream) {
  const float* x      = (const float*)d_in[0];
  const float* ln1_g  = (const float*)d_in[1];
  const float* ln1_b  = (const float*)d_in[2];
  const float* w_qkv  = (const float*)d_in[3];
  const float* w_proj = (const float*)d_in[4];
  const float* b_proj = (const float*)d_in[5];
  const float* ln2_g  = (const float*)d_in[6];
  const float* ln2_b  = (const float*)d_in[7];
  const float* w_fc1  = (const float*)d_in[8];
  const float* b_fc1  = (const float*)d_in[9];
  const float* w_fc2  = (const float*)d_in[10];
  const float* b_fc2  = (const float*)d_in[11];
  float* out = (float*)d_out;

  char* ws = (char*)d_ws;
  size_t off = 0;
  auto carve = [&](size_t bytes) -> char* {
    char* p = ws + off;
    off += (bytes + 255) & ~(size_t)255;
    return p;
  };
  __bf16* wqkv_b  = (__bf16*)carve((size_t)3 * CH * CH * 2);
  __bf16* wproj_b = (__bf16*)carve((size_t)CH * CH * 2);
  __bf16* wfc1_b  = (__bf16*)carve((size_t)HIDD * CH * 2);
  __bf16* wfc2_b  = (__bf16*)carve((size_t)CH * HIDD * 2);
  __bf16* xn1     = (__bf16*)carve((size_t)MTOK * CH * 2);
  __bf16* qb      = (__bf16*)carve((size_t)BH * SEQ * HD * 2);
  __bf16* kb      = (__bf16*)carve((size_t)BH * SEQ * HD * 2);
  __bf16* vtb     = (__bf16*)carve((size_t)BH * HD * NPAD * 2);
  __bf16* attno   = (__bf16*)carve((size_t)MTOK * CH * 2);
  float*  y1      = (float*) carve((size_t)MTOK * CH * 4);
  __bf16* xn2     = (__bf16*)carve((size_t)MTOK * CH * 2);
  __bf16* hbuf    = (__bf16*)carve((size_t)MTOK * HIDD * 2);

  // 1) weights -> bf16
  cast_bf16_kernel<<<2048, 256, 0, stream>>>(w_qkv,  wqkv_b,  (long)3 * CH * CH);
  cast_bf16_kernel<<<1024, 256, 0, stream>>>(w_proj, wproj_b, (long)CH * CH);
  cast_bf16_kernel<<<2048, 256, 0, stream>>>(w_fc1,  wfc1_b,  (long)HIDD * CH);
  cast_bf16_kernel<<<2048, 256, 0, stream>>>(w_fc2,  wfc2_b,  (long)CH * HIDD);
  // zero v^T (key padding region must not be NaN/inf)
  (void)hipMemsetAsync(vtb, 0, (size_t)BH * HD * NPAD * 2, stream);

  const int MT = (MTOK + 127) / 128;  // 145 row tiles

  // 2) LN1 -> bf16
  ln_bf16_kernel<<<MTOK, 256, 0, stream>>>(x, ln1_g, ln1_b, xn1);

  // 3) QKV GEMM (mode 2 scatter, q scaled by 1/8)
  gemm_bf16_kernel<<<dim3(3 * CH / 128, MT), 256, 0, stream>>>(
      xn1, wqkv_b, nullptr, nullptr, nullptr, nullptr, qb, kb, vtb,
      MTOK, 3 * CH, CH, 2);

  // 4) fused attention
  attn_kernel<<<dim3(37, BH), 128, 0, stream>>>(qb, kb, vtb, attno);

  // 5) proj GEMM + bias + residual(x) -> y1 (f32)
  gemm_bf16_kernel<<<dim3(CH / 128, MT), 256, 0, stream>>>(
      attno, wproj_b, b_proj, x, y1, nullptr, nullptr, nullptr, nullptr,
      MTOK, CH, CH, 0);

  // 6) LN2 -> bf16
  ln_bf16_kernel<<<MTOK, 256, 0, stream>>>(y1, ln2_g, ln2_b, xn2);

  // 7) fc1 GEMM + bias + GELU -> bf16
  gemm_bf16_kernel<<<dim3(HIDD / 128, MT), 256, 0, stream>>>(
      xn2, wfc1_b, b_fc1, nullptr, nullptr, hbuf, nullptr, nullptr, nullptr,
      MTOK, HIDD, CH, 1);

  // 8) fc2 GEMM + bias + residual(y1) -> out (f32)
  gemm_bf16_kernel<<<dim3(CH / 128, MT), 256, 0, stream>>>(
      hbuf, wfc2_b, b_fc2, y1, out, nullptr, nullptr, nullptr, nullptr,
      MTOK, CH, HIDD, 0);
}
